// Encoder_DRNN_2637109919757
// MI455X (gfx1250) — matrix operand order repaired and checked by reading, not compiled, not run on hardware
//
#include <hip/hip_runtime.h>
#include <hip/hip_bf16.h>

typedef __attribute__((ext_vector_type(16))) _Float16 v16h;
typedef __attribute__((ext_vector_type(8)))  float    v8f;

#define BB 4096
#define TT 50
#define HH 128
#define G3 384   // 3*H

// Load a 16x32 (MxK) f16 A-fragment chunk from an LDS [16][128] f16 tile.
// Contiguous A layout (matches B fragments): lane = (m | g<<4); elements 0-15 hold
// K=16g..16g+15 -> two adjacent 16B chunks per lane.
__device__ __forceinline__ v16h load_a_frag_lds(const _Float16* base, int m, int g, int ck) {
  union { v16h v; uint4 q[2]; } u;
  const _Float16* p = base + m * HH + ck * 32 + 16 * g;
  u.q[0] = *reinterpret_cast<const uint4*>(p);
  u.q[1] = *reinterpret_cast<const uint4*>(p + 8);
  return u.v;
}

// Persistent-tile dilated GRU layer. One workgroup = 16 batch rows, 24 waves,
// wave w owns gate columns [16w, 16w+16). Weights register-resident as WMMA
// B-fragments; hidden-state ring (4 slots) + gate buffers in LDS.
template<int RATE, bool FINAL>
__global__ __launch_bounds__(768, 1)
void gru_layer_kernel(const _Float16* __restrict__ Xin,   // [B,T,128] f16
                      const float* __restrict__ Wih,      // [384,128]
                      const float* __restrict__ Whh,      // [384,128]
                      const float* __restrict__ bih,      // [384]
                      const float* __restrict__ bhh,      // [384]
                      _Float16* __restrict__ Xout,        // [B,T,128] f16 (if !FINAL)
                      float* __restrict__ Fout,           // [B,T,128] f32 (if FINAL)
                      const float* __restrict__ maskv)    // [B] (if FINAL)
{
  __shared__ __align__(16) _Float16 Xstage[16 * HH];       // 4 KB
  __shared__ __align__(16) _Float16 Hring[4 * 16 * HH];    // 16 KB
  __shared__ float Ssum[16 * G3];                           // 24 KB
  __shared__ float Shn[16 * HH];                            // 8 KB

  const int tid  = threadIdx.x;
  const int w    = tid >> 5;          // wave 0..23
  const int lane = tid & 31;
  const int nl   = lane & 15;
  const int g    = lane >> 4;
  const int b0   = blockIdx.x * 16;
  const int col  = w * 16 + nl;       // 0..383

  // Wave-uniform scalar: is this wave an n-gate wave (columns 256..383)?
  // readfirstlane makes it an SGPR -> uniform s_cbranch instead of exec-mask
  // divergence around the per-timestep LDS scatter.
  const int is_n_gate = __builtin_amdgcn_readfirstlane((w >= 16) ? 1 : 0);

  // zero the hidden-state ring (initial state = 0 for first RATE steps)
  for (int e = tid; e < 4 * 16 * HH; e += 768) Hring[e] = (_Float16)0.f;

  // Preload register-resident B-fragments: B layout lane=(n | g<<4);
  // chunk ck covers K = 32*ck + 16*g .. +15 (16 contiguous values of row n).
  v16h wihf[4], whhf[4];
  {
    const float* wr = Wih + (size_t)col * HH;
    const float* hr = Whh + (size_t)col * HH;
#pragma unroll
    for (int ck = 0; ck < 4; ++ck) {
      union { v16h v; _Float16 h[16]; } a, b;
      const float* p0 = wr + ck * 32 + 16 * g;
      const float* p1 = hr + ck * 32 + 16 * g;
#pragma unroll
      for (int i = 0; i < 16; ++i) { a.h[i] = (_Float16)p0[i]; b.h[i] = (_Float16)p1[i]; }
      wihf[ck] = a.v;  whhf[ck] = b.v;
    }
  }
  const float bih_s = bih[col];
  const float bhh_s = bhh[col];

  // Loop-invariant LDS scatter indices (C layout: lane = n | g<<4, VGPR r -> m = r + 8g)
  int sidx[8];
#pragma unroll
  for (int r = 0; r < 8; ++r) sidx[r] = (r + 8 * g) * G3 + col;

  __syncthreads();

  for (int t = 0; t < TT; ++t) {
    const int cur = t & 3;
    const int prv = (t + (4 - RATE)) & 3;   // slot holding h[t-RATE] (zeros if t<RATE)

    // Stage X[b0..b0+15][t][:] into LDS (shared by all 24 waves' A-fragments)
    {
      const unsigned int* src =
          reinterpret_cast<const unsigned int*>(Xin + ((size_t)b0 * TT + t) * HH);
      unsigned int* dst = reinterpret_cast<unsigned int*>(Xstage);
      for (int e = tid; e < 16 * (HH / 2); e += 768) {
        int m = e >> 6, c = e & 63;
        dst[m * (HH / 2) + c] = src[(size_t)m * (TT * HH / 2) + c];
      }
      if (t + 1 < TT && tid < 16)
        __builtin_prefetch(Xin + (((size_t)(b0 + tid)) * TT + (t + 1)) * HH, 0, 1);
    }
    __syncthreads();

    // Load all 8 A-fragments first so DS latency overlaps the WMMA chain.
    const _Float16* hb = &Hring[prv * 16 * HH];
    v16h ax[4], ah[4];
#pragma unroll
    for (int ck = 0; ck < 4; ++ck) ax[ck] = load_a_frag_lds(Xstage, nl, g, ck);
#pragma unroll
    for (int ck = 0; ck < 4; ++ck) ah[ck] = load_a_frag_lds(hb, nl, g, ck);

    // gi = X_t * Wih^T ; gh = h[t-RATE] * Whh^T   (K=128 in 4 chunks of 32)
    v8f gi = {0.f, 0.f, 0.f, 0.f, 0.f, 0.f, 0.f, 0.f};
    v8f gh = gi;
#pragma unroll
    for (int ck = 0; ck < 4; ++ck) {
      gi = __builtin_amdgcn_wmma_f32_16x16x32_f16(false, ax[ck], false, wihf[ck],
                                                  (short)0, gi, false, false);
      gh = __builtin_amdgcn_wmma_f32_16x16x32_f16(false, ah[ck], false, whhf[ck],
                                                  (short)0, gh, false, false);
    }

    // Scatter C tiles to LDS gate buffers (uniform branch per wave).
    // r/z gates only need gi+gh; n gate needs gi and gh kept separate.
    if (!is_n_gate) {
#pragma unroll
      for (int r = 0; r < 8; ++r)
        Ssum[sidx[r]] = gi[r] + gh[r] + bih_s + bhh_s;
    } else {
#pragma unroll
      for (int r = 0; r < 8; ++r) {
        Ssum[sidx[r]]                      = gi[r] + bih_s;
        Shn[(r + 8 * g) * HH + (col - 256)] = gh[r] + bhh_s;
      }
    }
    __syncthreads();

    // Elementwise GRU update for the 16x128 tile
    for (int e = tid; e < 16 * HH; e += 768) {
      int m = e >> 7, j = e & 127;
      float rr  = 1.f / (1.f + __expf(-Ssum[m * G3 + j]));
      float zz  = 1.f / (1.f + __expf(-Ssum[m * G3 + HH + j]));
      float pre = Ssum[m * G3 + 2 * HH + j] + rr * Shn[m * HH + j];
      pre = fminf(fmaxf(pre, -15.f), 15.f);
      float ex = __expf(2.f * pre);
      float nn = (ex - 1.f) / (ex + 1.f);          // tanh
      float hp = (float)Hring[prv * 16 * HH + e];  // carry = h[t-RATE]
      float hv = (1.f - zz) * nn + zz * hp;
      Hring[cur * 16 * HH + e] = (_Float16)hv;
      if (FINAL) {
        Fout[(((size_t)(b0 + m)) * TT + t) * HH + j] = hv * maskv[b0 + m];
      } else {
        Xout[(((size_t)(b0 + m)) * TT + t) * HH + j] = (_Float16)hv;
      }
    }
    __syncthreads();
  }
}

// Gather embeddings to f16: X0[b,t,:] = emb[token[b,t], :]
__global__ __launch_bounds__(128)
void embed_kernel(const int* __restrict__ tok, const float* __restrict__ emb,
                  _Float16* __restrict__ X0) {
  const int bt = blockIdx.x;
  const int c  = threadIdx.x;
  const int tk = tok[bt];
  X0[(size_t)bt * HH + c] = (_Float16)emb[(size_t)tk * HH + c];
}

// sent_mask[b] = any(token[b, :] > 0)
__global__ __launch_bounds__(256)
void mask_kernel(const int* __restrict__ tok, float* __restrict__ maskv) {
  const int b = blockIdx.x * blockDim.x + threadIdx.x;
  if (b < BB) {
    int cnt = 0;
    for (int t = 0; t < TT; ++t) cnt += (tok[b * TT + t] > 0) ? 1 : 0;
    maskv[b] = (cnt > 0) ? 1.f : 0.f;
  }
}

extern "C" void kernel_launch(void* const* d_in, const int* in_sizes, int n_in,
                              void* d_out, int out_size, void* d_ws, size_t ws_size,
                              hipStream_t stream) {
  (void)in_sizes; (void)n_in; (void)out_size; (void)ws_size;

  const int*   tok  = (const int*)d_in[0];
  // d_in[1] = mask_input, d_in[2] = len_seq: reference recomputes -> unused
  const float* emb  = (const float*)d_in[3];
  const float* Wih0 = (const float*)d_in[4];
  const float* Whh0 = (const float*)d_in[5];
  const float* bih0 = (const float*)d_in[6];
  const float* bhh0 = (const float*)d_in[7];
  const float* Wih1 = (const float*)d_in[8];
  const float* Whh1 = (const float*)d_in[9];
  const float* bih1 = (const float*)d_in[10];
  const float* bhh1 = (const float*)d_in[11];
  const float* Wih2 = (const float*)d_in[12];
  const float* Whh2 = (const float*)d_in[13];
  const float* bih2 = (const float*)d_in[14];
  const float* bhh2 = (const float*)d_in[15];

  const size_t xbytes = (size_t)BB * TT * HH * sizeof(_Float16);  // ~52.4 MB
  char* ws = (char*)d_ws;
  _Float16* Xa    = (_Float16*)(ws);
  _Float16* Xb    = (_Float16*)(ws + xbytes);
  float*    maskf = (float*)(ws + 2 * xbytes);

  embed_kernel<<<BB * TT, 128, 0, stream>>>(tok, emb, Xa);
  mask_kernel<<<BB / 256, 256, 0, stream>>>(tok, maskf);

  // Layer 0: rate 1, reads Xa, writes Xb
  gru_layer_kernel<1, false><<<BB / 16, 768, 0, stream>>>(
      Xa, Wih0, Whh0, bih0, bhh0, Xb, nullptr, nullptr);
  // Layer 1: rate 2, reads Xb, writes Xa
  gru_layer_kernel<2, false><<<BB / 16, 768, 0, stream>>>(
      Xb, Wih1, Whh1, bih1, bhh1, Xa, nullptr, nullptr);
  // Layer 2: rate 4, reads Xa, writes masked f32 output
  gru_layer_kernel<4, true><<<BB / 16, 768, 0, stream>>>(
      Xa, Wih2, Whh2, bih2, bhh2, nullptr, (float*)d_out, maskf);
}
